// GeneralGNN_84739704750878
// MI455X (gfx1250) — matrix operand order repaired and checked by reading, not compiled, not run on hardware
//
#include <hip/hip_runtime.h>
#include <math.h>

#define NN 20000
#define NE 600000
#define HID 128
#define HEADS 4

typedef __attribute__((ext_vector_type(16))) __bf16 v16bf;
typedef __attribute__((ext_vector_type(8)))  float  v8f;

union ABits { unsigned int u32[8]; v16bf v; };

__device__ __forceinline__ unsigned short f32_to_bf16(float f) {
  union { float f; unsigned int u; } a; a.f = f;
  unsigned int u = a.u;
  u = (u + 0x7FFFu + ((u >> 16) & 1u)) >> 16;   // round-to-nearest-even
  return (unsigned short)u;
}

__device__ __forceinline__ void store4_bf16(unsigned short* dst, float4 v) {
  union { unsigned short u[4]; unsigned long long d; } t;
  t.u[0] = f32_to_bf16(v.x); t.u[1] = f32_to_bf16(v.y);
  t.u[2] = f32_to_bf16(v.z); t.u[3] = f32_to_bf16(v.w);
  *(unsigned long long*)dst = t.d;
}

__device__ __forceinline__ float gelu_f(float x) {
  return 0.5f * x * (1.0f + erff(x * 0.7071067811865475f));
}

__device__ __forceinline__ void atomicMaxF(float* addr, float val) {
  if (val >= 0.0f) atomicMax((int*)addr, __float_as_int(val));
  else             atomicMin((unsigned int*)addr, __float_as_uint(val));
}

// ---- WMMA fragment loaders (CDNA5 ISA 7.12.2 layouts, wave32) ----
// A: 16 x K row-major bf16 (ushort) in LDS; tile starting at column k0.
__device__ __forceinline__ v16bf load_a_frag(const unsigned short* A, int stride, int k0) {
  const int lane = threadIdx.x & 31;
  const int row = lane & 15, hh = lane >> 4;
  ABits r;
#pragma unroll
  for (int v = 0; v < 8; ++v) {
    int kb = k0 + ((v < 4) ? (2 * v) : (8 + 2 * v)) + 8 * hh; // K pairs 0..7/16..23 (+8 upper half)
    r.u32[v] = *(const unsigned int*)(A + row * stride + kb);
  }
  return r.v;
}

// B: fragment-major pre-swizzled weights. Fragment (kt,nt) holds 32 lanes x 16 bf16,
// each lane's 16 elements contiguous (32B) -> two global_load_b128.
__device__ __forceinline__ v16bf load_b_frag_sw(const unsigned short* SW, int NT, int kt, int nt) {
  const int lane = threadIdx.x & 31;
  return *(const v16bf*)(SW + (((size_t)(kt * NT + nt) * 32 + lane) << 4));
}

#define WMMA_BF16(a, b, c) \
  __builtin_amdgcn_wmma_f32_16x16x32_bf16(false, (a), false, (b), (short)0, (c), false, false)

// One 16-row MLP layer: O(16xN) = act(A(16xK) @ W(KxN) + bias), O stored bf16 in LDS.
template<int K, int N, bool GELU>
__device__ __forceinline__ void mlp_layer(const unsigned short* Alds,
                                          const unsigned short* SW,
                                          const float* bias,
                                          unsigned short* Olds) {
  const int lane = threadIdx.x & 31;
  const int col = lane & 15, hh = lane >> 4;
  constexpr int NT = N / 16;
  for (int nt = 0; nt < NT; ++nt) {
    v8f acc = {};
#pragma unroll
    for (int kt = 0; kt < K / 32; ++kt)
      acc = WMMA_BF16(load_a_frag(Alds, K, kt * 32), load_b_frag_sw(SW, NT, kt, nt), acc);
    float bv = bias[nt * 16 + col];
#pragma unroll
    for (int r = 0; r < 8; ++r) {
      float x = acc[r] + bv;
      if (GELU) x = gelu_f(x);
      Olds[(r + 8 * hh) * N + nt * 16 + col] = f32_to_bf16(x);
    }
  }
}

// ------------------------- utility kernels -------------------------
// Convert fp32 KxN weights (source has ic real cols, zero-pad to N) into
// fragment-major swizzled bf16 layout matching load_b_frag_sw.
__global__ void ggnn_cvt_sw(const float* __restrict__ s, unsigned short* __restrict__ d,
                            int K, int N, int ic) {
  int i = blockIdx.x * 256 + threadIdx.x;
  if (i >= K * N) return;
  int NT = N >> 4;
  int f = i >> 9;                 // fragment (512 elems)
  int r = i & 511;
  int L = r >> 4, e = r & 15;
  int kt = f / NT, nt = f - kt * NT;
  int sk = kt * 32 + ((L >> 4) << 3) + (e & 7) + ((e >> 3) << 4);
  int sn = nt * 16 + (L & 15);
  d[i] = (sn < ic) ? f32_to_bf16(s[(size_t)sk * ic + sn]) : (unsigned short)0;
}
__global__ void ggnn_pad_bias(const float* __restrict__ s, float* __restrict__ d, int ic, int oc) {
  int i = threadIdx.x;
  if (i < oc) d[i] = (i < ic) ? s[i] : 0.0f;
}
__global__ void ggnn_fill(float* __restrict__ p, float v, long n) {
  long i = (long)blockIdx.x * 256 + threadIdx.x;
  long stride = (long)gridDim.x * 256;
  for (; i < n; i += stride) p[i] = v;
}
__global__ void ggnn_bn_finalize(const float* __restrict__ s, const float* __restrict__ q,
                                 float* __restrict__ mean, float* __restrict__ rstd, float count) {
  int i = threadIdx.x;
  float m = s[i] / count;
  float v = q[i] / count - m * m;
  mean[i] = m;
  rstd[i] = rsqrtf(v + 1e-5f);
}

// ------------------- edge pass 1: Bias MLP + segment max -------------------
#define WPB 4
__global__ void __launch_bounds__(128)
ggnn_edge_bias(const float* __restrict__ hV, const float* __restrict__ hE,
               const int* __restrict__ eidx,
               const unsigned short* __restrict__ wbb1, const float* __restrict__ bb1b,
               const unsigned short* __restrict__ wbb2, const float* __restrict__ bb2b,
               const unsigned short* __restrict__ wbb3, const float* __restrict__ bb3bp,
               float* __restrict__ wlog, float* __restrict__ mmax) {
  __shared__ __attribute__((aligned(16))) unsigned short lds[WPB][16 * 384 + 16 * 128];
  __shared__ int sidx[WPB][32];                              // [0..15]=src, [16..31]=dst
  const int wave = threadIdx.x >> 5, lane = threadIdx.x & 31;
  const int t = blockIdx.x * WPB + wave;
  unsigned short* X = lds[wave];
  unsigned short* H1 = X + 16 * 384;
  const int e0 = t * 16;
  sidx[wave][lane] = eidx[(lane < 16 ? 0 : NE - 16) + e0 + lane];
  __syncthreads();
  const int* src = sidx[wave];
  const int* dst = sidx[wave] + 16;
  for (int i4 = lane; i4 < 16 * 96; i4 += 32) {              // float4 gather [hV[src]|hE|hV[dst]]
    int row = i4 / 96, c4 = i4 - row * 96;
    float4 v;
    if (c4 < 32)      v = ((const float4*)hV)[(size_t)src[row] * 32 + c4];
    else if (c4 < 64) v = ((const float4*)hE)[(size_t)(e0 + row) * 32 + (c4 - 32)];
    else              v = ((const float4*)hV)[(size_t)dst[row] * 32 + (c4 - 64)];
    store4_bf16(X + row * 384 + c4 * 4, v);
  }
  __syncthreads();
  mlp_layer<384, 128, true>(X, wbb1, bb1b, H1);
  __syncthreads();
  mlp_layer<128, 128, true>(H1, wbb2, bb2b, X);              // H2 overwrites head of X
  __syncthreads();
  {                                                          // logits = H2 @ bb3(pad16) + b
    const int col = lane & 15, hh = lane >> 4;
    v8f acc = {};
#pragma unroll
    for (int kt = 0; kt < 4; ++kt)
      acc = WMMA_BF16(load_a_frag(X, 128, kt * 32), load_b_frag_sw(wbb3, 1, kt, 0), acc);
    if (col < HEADS) {
      float bv = bb3bp[col];
#pragma unroll
      for (int r = 0; r < 8; ++r) {
        int row = r + 8 * hh;
        float x = acc[r] + bv;
        wlog[(size_t)(e0 + row) * HEADS + col] = x;
        atomicMaxF(&mmax[(size_t)src[row] * HEADS + col], x);
      }
    }
  }
}

// ---------- edge pass 2: Value MLP + softmax accumulate (denom, agg) ----------
__global__ void __launch_bounds__(128)
ggnn_edge_value(const float* __restrict__ hV, const float* __restrict__ hE,
                const int* __restrict__ eidx,
                const unsigned short* __restrict__ wv1, const float* __restrict__ wv1b,
                const unsigned short* __restrict__ wv2, const float* __restrict__ wv2b,
                const unsigned short* __restrict__ wv3, const float* __restrict__ wv3b,
                const float* __restrict__ wlog, const float* __restrict__ mmax,
                float* __restrict__ denom, float* __restrict__ agg) {
  __shared__ __attribute__((aligned(16))) unsigned short lds[WPB][16 * 256 + 16 * 128];
  __shared__ float ewl[WPB][64];
  __shared__ int sidx[WPB][32];
  const int wave = threadIdx.x >> 5, lane = threadIdx.x & 31;
  const int t = blockIdx.x * WPB + wave;
  unsigned short* X = lds[wave];
  unsigned short* H1 = X + 16 * 256;
  const int e0 = t * 16;
  sidx[wave][lane] = eidx[(lane < 16 ? 0 : NE - 16) + e0 + lane];
  __syncthreads();
  const int* src = sidx[wave];
  const int* dst = sidx[wave] + 16;
  for (int i4 = lane; i4 < 16 * 64; i4 += 32) {              // float4 gather [hE | hV[dst]]
    int row = i4 >> 6, c4 = i4 & 63;
    float4 v = (c4 < 32) ? ((const float4*)hE)[(size_t)(e0 + row) * 32 + c4]
                         : ((const float4*)hV)[(size_t)dst[row] * 32 + (c4 - 32)];
    store4_bf16(X + row * 256 + c4 * 4, v);
  }
  for (int idx = lane; idx < 64; idx += 32) {                // ew = exp(w - m[src]); denom += ew
    int row = idx >> 2, h = idx & 3;
    int e = e0 + row, s = src[row];
    float ew = expf(wlog[(size_t)e * HEADS + h] - mmax[(size_t)s * HEADS + h]);
    ewl[wave][idx] = ew;
    atomicAdd(&denom[(size_t)s * HEADS + h], ew);
  }
  __syncthreads();
  mlp_layer<256, 128, true>(X, wv1, wv1b, H1);
  __syncthreads();
  mlp_layer<128, 128, true>(H1, wv2, wv2b, X);
  __syncthreads();
  const int col = lane & 15, hh = lane >> 4;
  for (int nt = 0; nt < 8; ++nt) {                           // V = H2 @ wv3 + b; agg += ew*V
    v8f acc = {};
#pragma unroll
    for (int kt = 0; kt < 4; ++kt)
      acc = WMMA_BF16(load_a_frag(X, 128, kt * 32), load_b_frag_sw(wv3, 8, kt, nt), acc);
    float bv = wv3b[nt * 16 + col];
    int head = (nt * 16 + col) >> 5;
#pragma unroll
    for (int r = 0; r < 8; ++r) {
      int row = r + 8 * hh;
      float v = (acc[r] + bv) * ewl[wave][row * 4 + head];
      atomicAdd(&agg[(size_t)src[row] * 128 + nt * 16 + col], v);
    }
  }
}

// ---------------- node pass 1: dh = (agg/denom) @ W_O; x0 = hV + dh; BN0 stats ----------------
__global__ void __launch_bounds__(128)
ggnn_node_wo(const float* __restrict__ hV, const float* __restrict__ agg,
             const float* __restrict__ denom, const unsigned short* __restrict__ wo,
             float* __restrict__ x, float* __restrict__ s0, float* __restrict__ q0) {
  __shared__ __attribute__((aligned(16))) unsigned short lds[WPB][16 * 128];
  const int wave = threadIdx.x >> 5, lane = threadIdx.x & 31;
  const int t = blockIdx.x * WPB + wave;
  const bool active = t < (NN / 16);
  const int nb = t * 16;
  if (active) {
    for (int i4 = lane; i4 < 512; i4 += 32) {                // 16 nodes x 32 float4
      int row = i4 >> 5, c4 = i4 & 31, node = nb + row;
      float d = denom[(size_t)node * HEADS + (c4 >> 3)];
      float inv = d > 0.0f ? 1.0f / d : 0.0f;
      float4 a = ((const float4*)agg)[(size_t)node * 32 + c4];
      float4 v = {a.x * inv, a.y * inv, a.z * inv, a.w * inv};
      store4_bf16(lds[wave] + row * 128 + c4 * 4, v);
    }
  }
  __syncthreads();
  if (active) {
    const int col = lane & 15, hh = lane >> 4;
    for (int nt = 0; nt < 8; ++nt) {
      v8f acc = {};
#pragma unroll
      for (int kt = 0; kt < 4; ++kt)
        acc = WMMA_BF16(load_a_frag(lds[wave], 128, kt * 32), load_b_frag_sw(wo, 8, kt, nt), acc);
      float sum = 0.f, sq = 0.f;
#pragma unroll
      for (int r = 0; r < 8; ++r) {
        int node = nb + r + 8 * hh;
        float xv = hV[(size_t)node * 128 + nt * 16 + col] + acc[r];
        x[(size_t)node * 128 + nt * 16 + col] = xv;
        sum += xv; sq += xv * xv;
      }
      sum += __shfl_xor(sum, 16, 32);
      sq  += __shfl_xor(sq, 16, 32);
      if (hh == 0) { atomicAdd(&s0[nt * 16 + col], sum); atomicAdd(&q0[nt * 16 + col], sq); }
    }
  }
}

// ---------------- node pass 2: FFN 128->512->128 + residual + BN1 stats ----------------
__global__ void __launch_bounds__(64)
ggnn_node_ffn(const float* __restrict__ x,
              const float* __restrict__ mean0, const float* __restrict__ rstd0,
              const float* __restrict__ g0, const float* __restrict__ b0,
              const unsigned short* __restrict__ wd1, const float* __restrict__ d1b,
              const unsigned short* __restrict__ wd2, const float* __restrict__ d2b,
              float* __restrict__ xout, float* __restrict__ s1, float* __restrict__ q1) {
  __shared__ __attribute__((aligned(16))) unsigned short h1l[2][16 * 128];
  __shared__ __attribute__((aligned(16))) unsigned short Tl[2][16 * 256];
  const int wave = threadIdx.x >> 5, lane = threadIdx.x & 31;
  const int t = blockIdx.x * 2 + wave;
  const bool active = t < (NN / 16);
  const int nb = t * 16;
  const int col = lane & 15, hh = lane >> 4;
  if (active) {
    for (int i4 = lane; i4 < 512; i4 += 32) {
      int row = i4 >> 5, c4 = i4 & 31, c = c4 * 4;
      float4 xv = ((const float4*)x)[(size_t)(nb + row) * 32 + c4];
      float4 h;
      h.x = g0[c + 0] * (xv.x - mean0[c + 0]) * rstd0[c + 0] + b0[c + 0];
      h.y = g0[c + 1] * (xv.y - mean0[c + 1]) * rstd0[c + 1] + b0[c + 1];
      h.z = g0[c + 2] * (xv.z - mean0[c + 2]) * rstd0[c + 2] + b0[c + 2];
      h.w = g0[c + 3] * (xv.w - mean0[c + 3]) * rstd0[c + 3] + b0[c + 3];
      store4_bf16(h1l[wave] + row * 128 + c, h);
    }
  }
  __syncthreads();
  v8f dh[8];
#pragma unroll
  for (int j = 0; j < 8; ++j) dh[j] = (v8f){};
  for (int chunk = 0; chunk < 2; ++chunk) {
    if (active) {                                            // T = gelu(h1 @ d1[:,chunk])
      for (int nt = 0; nt < 16; ++nt) {
        v8f acc = {};
#pragma unroll
        for (int kt = 0; kt < 4; ++kt)
          acc = WMMA_BF16(load_a_frag(h1l[wave], 128, kt * 32),
                          load_b_frag_sw(wd1, 32, kt, chunk * 16 + nt), acc);
        float bv = d1b[chunk * 256 + nt * 16 + col];
#pragma unroll
        for (int r = 0; r < 8; ++r)
          Tl[wave][(r + 8 * hh) * 256 + nt * 16 + col] = f32_to_bf16(gelu_f(acc[r] + bv));
      }
    }
    __syncthreads();
    if (active) {                                            // dh += T @ d2[chunk,:]
#pragma unroll
      for (int j = 0; j < 8; ++j)
        for (int kt = 0; kt < 8; ++kt)
          dh[j] = WMMA_BF16(load_a_frag(Tl[wave], 256, kt * 32),
                            load_b_frag_sw(wd2, 8, chunk * 8 + kt, j), dh[j]);
    }
    __syncthreads();
  }
  if (active) {
#pragma unroll
    for (int j = 0; j < 8; ++j) {
      int n0 = j * 16;
      float bv = d2b[n0 + col];
      float sum = 0.f, sq = 0.f;
#pragma unroll
      for (int r = 0; r < 8; ++r) {
        int node = nb + r + 8 * hh;
        size_t o = (size_t)node * 128 + n0 + col;
        float h1f = g0[n0 + col] * (x[o] - mean0[n0 + col]) * rstd0[n0 + col] + b0[n0 + col];
        float xv = h1f + dh[j][r] + bv;
        xout[o] = xv;
        sum += xv; sq += xv * xv;
      }
      sum += __shfl_xor(sum, 16, 32);
      sq  += __shfl_xor(sq, 16, 32);
      if (hh == 0) { atomicAdd(&s1[n0 + col], sum); atomicAdd(&q1[n0 + col], sq); }
    }
  }
}

__global__ void ggnn_hv_final(const float* __restrict__ x,
                              const float* __restrict__ mean1, const float* __restrict__ rstd1,
                              const float* __restrict__ g, const float* __restrict__ b,
                              float* __restrict__ hvf, float* __restrict__ out) {
  long i4 = (long)blockIdx.x * 256 + threadIdx.x;
  if (i4 < (long)NN * 32) {
    int c = (i4 & 31) * 4;
    float4 xv = ((const float4*)x)[i4];
    float4 v;
    v.x = g[c + 0] * (xv.x - mean1[c + 0]) * rstd1[c + 0] + b[c + 0];
    v.y = g[c + 1] * (xv.y - mean1[c + 1]) * rstd1[c + 1] + b[c + 1];
    v.z = g[c + 2] * (xv.z - mean1[c + 2]) * rstd1[c + 2] + b[c + 2];
    v.w = g[c + 3] * (xv.w - mean1[c + 3]) * rstd1[c + 3] + b[c + 3];
    ((float4*)hvf)[i4] = v;
    ((float4*)out)[i4] = v;
  }
}

// ---------------- edge pass 3: EdgeMLP + residual -> d_out (pre-norm) + BN stats ----------------
__global__ void __launch_bounds__(128)
ggnn_edge_mlp(const float* __restrict__ hvf, const float* __restrict__ hE,
              const int* __restrict__ eidx,
              const unsigned short* __restrict__ we1, const float* __restrict__ e1b,
              const unsigned short* __restrict__ we2, const float* __restrict__ e2b,
              const unsigned short* __restrict__ we3, const float* __restrict__ e3b,
              float* __restrict__ outE, float* __restrict__ sE, float* __restrict__ qE) {
  __shared__ __attribute__((aligned(16))) unsigned short lds[WPB][16 * 384 + 16 * 128];
  __shared__ int sidx[WPB][32];
  const int wave = threadIdx.x >> 5, lane = threadIdx.x & 31;
  const int t = blockIdx.x * WPB + wave;
  unsigned short* X = lds[wave];
  unsigned short* H1 = X + 16 * 384;
  const int e0 = t * 16;
  sidx[wave][lane] = eidx[(lane < 16 ? 0 : NE - 16) + e0 + lane];
  __syncthreads();
  const int* src = sidx[wave];
  const int* dst = sidx[wave] + 16;
  for (int i4 = lane; i4 < 16 * 96; i4 += 32) {
    int row = i4 / 96, c4 = i4 - row * 96;
    float4 v;
    if (c4 < 32)      v = ((const float4*)hvf)[(size_t)src[row] * 32 + c4];
    else if (c4 < 64) v = ((const float4*)hE)[(size_t)(e0 + row) * 32 + (c4 - 32)];
    else              v = ((const float4*)hvf)[(size_t)dst[row] * 32 + (c4 - 64)];
    store4_bf16(X + row * 384 + c4 * 4, v);
  }
  __syncthreads();
  mlp_layer<384, 128, true>(X, we1, e1b, H1);
  __syncthreads();
  mlp_layer<128, 128, true>(H1, we2, e2b, X);
  __syncthreads();
  const int col = lane & 15, hh = lane >> 4;
  for (int nt = 0; nt < 8; ++nt) {
    v8f acc = {};
#pragma unroll
    for (int kt = 0; kt < 4; ++kt)
      acc = WMMA_BF16(load_a_frag(X, 128, kt * 32), load_b_frag_sw(we3, 8, kt, nt), acc);
    float bv = e3b[nt * 16 + col];
    float sum = 0.f, sq = 0.f;
#pragma unroll
    for (int r = 0; r < 8; ++r) {
      int e = e0 + r + 8 * hh;
      float xv = hE[(size_t)e * 128 + nt * 16 + col] + acc[r] + bv;
      outE[(size_t)e * 128 + nt * 16 + col] = xv;
      sum += xv; sq += xv * xv;
    }
    sum += __shfl_xor(sum, 16, 32);
    sq  += __shfl_xor(sq, 16, 32);
    if (hh == 0) { atomicAdd(&sE[nt * 16 + col], sum); atomicAdd(&qE[nt * 16 + col], sq); }
  }
}

__global__ void ggnn_he_norm(float* __restrict__ outE,
                             const float* __restrict__ meanE, const float* __restrict__ rstdE,
                             const float* __restrict__ g, const float* __restrict__ b) {
  long i4 = (long)blockIdx.x * 256 + threadIdx.x;
  long stride = (long)gridDim.x * 256;
  for (; i4 < (long)NE * 32; i4 += stride) {
    int c = (i4 & 31) * 4;
    float4 xv = ((const float4*)outE)[i4];
    float4 v;
    v.x = g[c + 0] * (xv.x - meanE[c + 0]) * rstdE[c + 0] + b[c + 0];
    v.y = g[c + 1] * (xv.y - meanE[c + 1]) * rstdE[c + 1] + b[c + 1];
    v.z = g[c + 2] * (xv.z - meanE[c + 2]) * rstdE[c + 2] + b[c + 2];
    v.w = g[c + 3] * (xv.w - meanE[c + 3]) * rstdE[c + 3] + b[c + 3];
    ((float4*)outE)[i4] = v;
  }
}

// ------------------------------- host launcher -------------------------------
extern "C" void kernel_launch(void* const* d_in, const int* in_sizes, int n_in,
                              void* d_out, int out_size, void* d_ws, size_t ws_size,
                              hipStream_t stream) {
  const float* hV   = (const float*)d_in[0];
  const float* hE   = (const float*)d_in[1];
  const int*   eidx = (const int*)d_in[2];
  const float* wv1w = (const float*)d_in[4];  const float* wv1b = (const float*)d_in[5];
  const float* wv2w = (const float*)d_in[6];  const float* wv2b = (const float*)d_in[7];
  const float* wv3w = (const float*)d_in[8];  const float* wv3b = (const float*)d_in[9];
  const float* bb1w = (const float*)d_in[10]; const float* bb1b = (const float*)d_in[11];
  const float* bb2w = (const float*)d_in[12]; const float* bb2b = (const float*)d_in[13];
  const float* bb3w = (const float*)d_in[14]; const float* bb3b = (const float*)d_in[15];
  const float* wow  = (const float*)d_in[16];
  const float* n0g  = (const float*)d_in[17]; const float* n0b  = (const float*)d_in[18];
  const float* n1g  = (const float*)d_in[19]; const float* n1b  = (const float*)d_in[20];
  const float* d1w  = (const float*)d_in[21]; const float* d1b  = (const float*)d_in[22];
  const float* d2w  = (const float*)d_in[23]; const float* d2b  = (const float*)d_in[24];
  const float* e1w  = (const float*)d_in[25]; const float* e1b  = (const float*)d_in[26];
  const float* e2w  = (const float*)d_in[27]; const float* e2b  = (const float*)d_in[28];
  const float* e3w  = (const float*)d_in[29]; const float* e3b  = (const float*)d_in[30];
  const float* eng  = (const float*)d_in[31]; const float* enb  = (const float*)d_in[32];

  char* base = (char*)d_ws;
  size_t off = 0;
  auto alloc = [&](size_t bytes) -> char* {
    char* p = base + off; off = (off + bytes + 255) & ~(size_t)255; return p;
  };
  unsigned short* WBB1 = (unsigned short*)alloc(384 * 128 * 2);
  unsigned short* WBB2 = (unsigned short*)alloc(128 * 128 * 2);
  unsigned short* WBB3 = (unsigned short*)alloc(128 * 16 * 2);
  unsigned short* WWV1 = (unsigned short*)alloc(256 * 128 * 2);
  unsigned short* WWV2 = (unsigned short*)alloc(128 * 128 * 2);
  unsigned short* WWV3 = (unsigned short*)alloc(128 * 128 * 2);
  unsigned short* WWO  = (unsigned short*)alloc(128 * 128 * 2);
  unsigned short* WD1  = (unsigned short*)alloc(128 * 512 * 2);
  unsigned short* WD2  = (unsigned short*)alloc(512 * 128 * 2);
  unsigned short* WE1  = (unsigned short*)alloc(384 * 128 * 2);
  unsigned short* WE2  = (unsigned short*)alloc(128 * 128 * 2);
  unsigned short* WE3  = (unsigned short*)alloc(128 * 128 * 2);
  float* BB3BP = (float*)alloc(16 * 4);
  float* STATS = (float*)alloc(12 * 128 * 4);
  float* M     = (float*)alloc((size_t)NN * HEADS * 4);
  float* DEN   = (float*)alloc((size_t)NN * HEADS * 4);
  float* AGG   = (float*)alloc((size_t)NN * 128 * 4);
  float* X     = (float*)alloc((size_t)NN * 128 * 4);
  float* HVF   = (float*)alloc((size_t)NN * 128 * 4);
  float* WLOG  = (float*)alloc((size_t)NE * HEADS * 4);

  float* s0 = STATS;        float* q0 = STATS + 128;
  float* s1 = STATS + 256;  float* q1 = STATS + 384;
  float* sE = STATS + 512;  float* qE = STATS + 640;
  float* mean0 = STATS + 768;  float* rstd0 = STATS + 896;
  float* mean1 = STATS + 1024; float* rstd1 = STATS + 1152;
  float* meanE = STATS + 1280; float* rstdE = STATS + 1408;

  auto cvtsw = [&](const float* s, unsigned short* d, int K, int N, int ic) {
    ggnn_cvt_sw<<<(K * N + 255) / 256, 256, 0, stream>>>(s, d, K, N, ic);
  };
  cvtsw(bb1w, WBB1, 384, 128, 128); cvtsw(bb2w, WBB2, 128, 128, 128);
  cvtsw(bb3w, WBB3, 128, 16, 4);
  cvtsw(wv1w, WWV1, 256, 128, 128); cvtsw(wv2w, WWV2, 128, 128, 128);
  cvtsw(wv3w, WWV3, 128, 128, 128);
  cvtsw(wow,  WWO,  128, 128, 128);
  cvtsw(d1w,  WD1,  128, 512, 512); cvtsw(d2w, WD2, 512, 128, 128);
  cvtsw(e1w,  WE1,  384, 128, 128); cvtsw(e2w, WE2, 128, 128, 128);
  cvtsw(e3w,  WE3,  128, 128, 128);
  ggnn_pad_bias<<<1, 16, 0, stream>>>(bb3b, BB3BP, 4, 16);

  ggnn_fill<<<256, 256, 0, stream>>>(M, -3.0e38f, (long)NN * HEADS);
  ggnn_fill<<<256, 256, 0, stream>>>(DEN, 0.0f, (long)NN * HEADS);
  ggnn_fill<<<1024, 256, 0, stream>>>(AGG, 0.0f, (long)NN * 128);
  ggnn_fill<<<4, 256, 0, stream>>>(STATS, 0.0f, 768);

  const int edgeBlocks = NE / 16 / WPB;  // 9375
  ggnn_edge_bias<<<edgeBlocks, 128, 0, stream>>>(hV, hE, eidx, WBB1, bb1b, WBB2, bb2b,
                                                 WBB3, BB3BP, WLOG, M);
  ggnn_edge_value<<<edgeBlocks, 128, 0, stream>>>(hV, hE, eidx, WWV1, wv1b, WWV2, wv2b,
                                                  WWV3, wv3b, WLOG, M, DEN, AGG);

  const int nodeTiles = NN / 16;  // 1250
  ggnn_node_wo<<<(nodeTiles + WPB - 1) / WPB, 128, 0, stream>>>(hV, AGG, DEN, WWO, X, s0, q0);
  ggnn_bn_finalize<<<1, 128, 0, stream>>>(s0, q0, mean0, rstd0, (float)NN);
  ggnn_node_ffn<<<(nodeTiles + 1) / 2, 64, 0, stream>>>(X, mean0, rstd0, n0g, n0b,
                                                        WD1, d1b, WD2, d2b, X, s1, q1);
  ggnn_bn_finalize<<<1, 128, 0, stream>>>(s1, q1, mean1, rstd1, (float)NN);

  float* outV = (float*)d_out;
  float* outE = outV + (size_t)NN * 128;
  ggnn_hv_final<<<((NN * 32) + 255) / 256, 256, 0, stream>>>(X, mean1, rstd1, n1g, n1b,
                                                             HVF, outV);
  ggnn_edge_mlp<<<edgeBlocks, 128, 0, stream>>>(HVF, hE, eidx, WE1, e1b, WE2, e2b,
                                                WE3, e3b, outE, sE, qE);
  ggnn_bn_finalize<<<1, 128, 0, stream>>>(sE, qE, meanE, rstdE, (float)NE);
  ggnn_he_norm<<<4096, 256, 0, stream>>>(outE, meanE, rstdE, eng, enb);
  (void)in_sizes; (void)n_in; (void)out_size; (void)ws_size;
}
